// InferDecoder_87565793231421
// MI455X (gfx1250) — matrix operand order repaired and checked by reading, not compiled, not run on hardware
//
#include <hip/hip_runtime.h>
#include <hip/hip_bf16.h>
#include <stdint.h>

#define B_ 32
#define H_ 1024
#define S_ 128
#define V_ 32000
#define T_ 64

typedef __attribute__((ext_vector_type(16))) __bf16 v16bf;
typedef __attribute__((ext_vector_type(8)))  float  v8f;

// ---------- helpers ----------

__device__ inline unsigned short f2bf(float f) {
  unsigned u = __float_as_uint(f);
  u += 0x7fffu + ((u >> 16) & 1u);          // round-to-nearest-even
  return (unsigned short)(u >> 16);
}

// A-fragment (16x32 bf16, wave32) lane/element layout per CDNA5 ISA 7.12.2:
// lanes 0-15 hold K=0..7 (v0..3) and 16..23 (v4..7); lanes 16-31 hold K=8..15, 24..31.
__device__ __host__ inline int a_sw_index(int m, int kk) {
  int kt    = kk >> 5;
  int ko    = kk & 31;
  int half  = (ko >> 3) & 1;
  int e     = (ko & 7) | ((ko >> 4) << 3);
  int lane  = (m & 15) | (half << 4);
  int mtile = m >> 4;
  return (((kt * 2 + mtile) * 32 + lane) << 4) + e;
}

__device__ inline v16bf load_frag(const unsigned short* p, int lane) {
  return *(const v16bf*)(p + (lane << 4));   // 32B aligned by construction
}

__device__ inline v8f wmma_bf16(v16bf a, v16bf b, v8f c) {
  // v_wmma_f32_16x16x32_bf16
  return __builtin_amdgcn_wmma_f32_16x16x32_bf16(false, a, false, b, (short)0, c,
                                                 false, false);
}

__device__ inline float sigf(float x) { return 1.f / (1.f + __expf(-x)); }

// ---------- one-time weight conversion: fp32 row-major -> bf16 WMMA-B-swizzled ----------
// B-fragment (32x16): lane = (n&15) | (((k%32)/8)&1)<<4 ; element e = (k%8) | ((k%32)/16)<<3.
__global__ __launch_bounds__(256) void swizzle_w_kernel(const float* __restrict__ src,
                                                        unsigned short* __restrict__ dst,
                                                        int K, int N) {
  size_t total  = (size_t)K * N;
  size_t stride = (size_t)gridDim.x * blockDim.x;
  int NT = N >> 4;
  for (size_t i = (size_t)blockIdx.x * blockDim.x + threadIdx.x; i < total; i += stride) {
    int    e    = (int)(i & 15);
    int    lane = (int)((i >> 4) & 31);
    size_t rest = i >> 9;
    int    nt   = (int)(rest % NT);
    int    kt   = (int)(rest / NT);
    int    k    = (kt << 5) + ((e >> 3) << 4) + ((lane >> 4) << 3) + (e & 7);
    int    n    = (nt << 4) + (lane & 15);
    dst[i] = f2bf(src[(size_t)k * N + n]);
  }
}

// ---------- state init ----------
__global__ __launch_bounds__(256) void init_cell_kernel(const float* __restrict__ h_in,
                                                        const float* __restrict__ c_in,
                                                        float* __restrict__ h,
                                                        float* __restrict__ c,
                                                        unsigned short* __restrict__ h_sw) {
  int idx = blockIdx.x * 256 + threadIdx.x;          // 32*1024
  int m = idx >> 10, k = idx & 1023;
  float hv = h_in[idx];
  h[idx] = hv;
  c[idx] = c_in[idx];
  h_sw[a_sw_index(m, k)] = f2bf(hv);
}

__global__ void init_token_kernel(int* token, int* done, int* flags) {
  int b = threadIdx.x;
  if (b < B_) { token[b] = 1; done[b] = 0; }         // START_TGT = 1
  if (b == 0) flags[0] = 0;                          // all_done = false
}

// ---------- embedding gather -> bf16 A-swizzled ----------
__global__ __launch_bounds__(256) void embed_kernel(const float* __restrict__ emb,
                                                    const int* __restrict__ token,
                                                    unsigned short* __restrict__ x_sw) {
  int idx = blockIdx.x * 256 + threadIdx.x;          // 32*1024
  int m = idx >> 10, k = idx & 1023;
  x_sw[a_sw_index(m, k)] = f2bf(emb[(size_t)token[m] * H_ + k]);
}

// ---------- fused LSTM cell: z = x@W + h@U + b, gates, c/h update ----------
// grid = 64 blocks (16 h-cols each), block = 256 threads = 8 waves.
// wave w: mtile = w&1, gate = w>>1 -> one 16x16 WMMA tile, 64 K-iterations.
__global__ __launch_bounds__(256) void lstm_cell_kernel(
    const unsigned short* __restrict__ A1,   // input act, K=1024, A-swizzled bf16
    const unsigned short* __restrict__ Wsw,  // 1024x4096 B-swizzled bf16
    const unsigned short* __restrict__ A2,   // previous h (this cell), A-swizzled bf16
    const unsigned short* __restrict__ Usw,  // 1024x4096 B-swizzled bf16
    const float* __restrict__ bias,          // 4096
    float* __restrict__ c, float* __restrict__ h,
    unsigned short* __restrict__ h_sw_out)   // next h, A-swizzled (double-buffered)
{
  __shared__ float zl[32][64];
  const int lane  = threadIdx.x & 31;
  const int wv    = threadIdx.x >> 5;
  const int mtile = wv & 1;
  const int gate  = wv >> 1;                 // 0..3
  const int n0    = blockIdx.x << 4;         // h-column base
  const int NT    = (4 * H_) >> 4;           // 256 column tiles in gate space
  const int nt    = (gate << 6) + (n0 >> 4); // gate*64 + block

  v8f acc = {};
  for (int kt = 0; kt < 32; ++kt) {          // x @ W
    v16bf a = load_frag(A1 + ((kt * 2 + mtile) << 9), lane);
    const unsigned short* bp = Wsw + (((size_t)kt * NT + nt) << 9);
    __builtin_prefetch(bp + ((size_t)NT << 9), 0, 1);   // next kt fragment
    v16bf b = load_frag(bp, lane);
    acc = wmma_bf16(a, b, acc);
  }
  for (int kt = 0; kt < 32; ++kt) {          // h @ U
    v16bf a = load_frag(A2 + ((kt * 2 + mtile) << 9), lane);
    const unsigned short* bp = Usw + (((size_t)kt * NT + nt) << 9);
    __builtin_prefetch(bp + ((size_t)NT << 9), 0, 1);
    v16bf b = load_frag(bp, lane);
    acc = wmma_bf16(a, b, acc);
  }

  // C/D layout: lane n = lane&15, row = r + 8*(lane>>4) (+16*mtile)
  int nloc  = lane & 15;
  int mbase = ((lane >> 4) << 3) + (mtile << 4);
#pragma unroll
  for (int r = 0; r < 8; ++r) zl[mbase + r][(gate << 4) + nloc] = acc[r];
  __syncthreads();

  for (int idx = threadIdx.x; idx < 512; idx += 256) {
    int m = idx >> 4, j = idx & 15;
    int n = n0 + j;
    float iv = sigf(zl[m][j]      + bias[n]);
    float fv = sigf(zl[m][16 + j] + bias[H_ + n]);
    float gv = tanhf(zl[m][32 + j] + bias[2 * H_ + n]);
    float ov = sigf(zl[m][48 + j] + bias[3 * H_ + n]);
    float cn = fv * c[m * H_ + n] + iv * gv;
    float hn = ov * tanhf(cn);
    c[m * H_ + n] = cn;
    h[m * H_ + n] = hn;
    h_sw_out[a_sw_index(m, n)] = f2bf(hn);
  }
}

// ---------- attention ----------
__global__ __launch_bounds__(256) void attn_scores_kernel(const float* __restrict__ h4,
                                                          const float* __restrict__ enc,
                                                          const unsigned char* __restrict__ mask,
                                                          float* __restrict__ scores) {
  int pair = blockIdx.x * 8 + (threadIdx.x >> 5);    // 4096 (b,s) pairs
  int lane = threadIdx.x & 31;
  int b = pair >> 7, s = pair & 127;
  const float* e = enc + ((size_t)b * S_ + s) * H_;
  const float* hp = h4 + b * H_;
  float sum = 0.f;
  for (int k = lane; k < H_; k += 32) sum += e[k] * hp[k];
  for (int off = 16; off; off >>= 1) sum += __shfl_xor(sum, off, 32);
  if (lane == 0) scores[b * S_ + s] = mask[b * S_ + s] ? sum : -1e9f;
}

__global__ __launch_bounds__(128) void attn_softmax_kernel(const float* __restrict__ scores,
                                                           float* __restrict__ w) {
  __shared__ float red[128];
  int b = blockIdx.x, t = threadIdx.x;
  float v = scores[b * S_ + t];
  red[t] = v; __syncthreads();
  for (int off = 64; off; off >>= 1) { if (t < off) red[t] = fmaxf(red[t], red[t + off]); __syncthreads(); }
  float mx = red[0]; __syncthreads();
  float e = __expf(v - mx);
  red[t] = e; __syncthreads();
  for (int off = 64; off; off >>= 1) { if (t < off) red[t] += red[t + off]; __syncthreads(); }
  w[b * S_ + t] = e / red[0];
}

__global__ __launch_bounds__(256) void context_cat_kernel(const float* __restrict__ w,
                                                          const float* __restrict__ enc,
                                                          const float* __restrict__ h4,
                                                          unsigned short* __restrict__ cat_sw) {
  int idx = blockIdx.x * 256 + threadIdx.x;          // 32*1024
  int b = idx >> 10, hc = idx & 1023;
  const float* e  = enc + (size_t)b * S_ * H_ + hc;
  const float* wr = w + b * S_;
  float s = 0.f;
  for (int j = 0; j < S_; ++j) s += wr[j] * e[(size_t)j * H_];
  cat_sw[a_sw_index(b, hc)]      = f2bf(s);          // context
  cat_sw[a_sw_index(b, H_ + hc)] = f2bf(h4[idx]);    // h4
}

// ---------- fc0: (32x2048)@(2048x1024) + b, tanh, -> A-swizzled bf16 ----------
__global__ __launch_bounds__(256) void fc0_kernel(const unsigned short* __restrict__ A,
                                                  const unsigned short* __restrict__ Bm,
                                                  const float* __restrict__ bias,
                                                  unsigned short* __restrict__ att_sw) {
  const int lane  = threadIdx.x & 31;
  const int wv    = threadIdx.x >> 5;
  const int mtile = wv & 1;
  const int ntl   = wv >> 1;
  const int NT    = H_ >> 4;                 // 64
  const int nt    = blockIdx.x * 4 + ntl;    // 16 blocks * 4 tiles
  v8f acc = {};
  for (int kt = 0; kt < 64; ++kt) {          // K = 2048
    v16bf a = load_frag(A + ((kt * 2 + mtile) << 9), lane);
    const unsigned short* bp = Bm + (((size_t)kt * NT + nt) << 9);
    __builtin_prefetch(bp + ((size_t)NT << 9), 0, 1);
    v16bf b = load_frag(bp, lane);
    acc = wmma_bf16(a, b, acc);
  }
  int n  = (nt << 4) + (lane & 15);
  int mb = ((lane >> 4) << 3) + (mtile << 4);
#pragma unroll
  for (int r = 0; r < 8; ++r) {
    float v = tanhf(acc[r] + bias[n]);
    att_sw[a_sw_index(mb + r, n)] = f2bf(v);
  }
}

// ---------- fc1: (32x1024)@(1024x32000) + b -> logits fp32 ----------
__global__ __launch_bounds__(256) void fc1_kernel(const unsigned short* __restrict__ A,
                                                  const unsigned short* __restrict__ Bm,
                                                  const float* __restrict__ bias,
                                                  float* __restrict__ logits) {
  const int lane  = threadIdx.x & 31;
  const int wv    = threadIdx.x >> 5;
  const int mtile = wv & 1;
  const int ntl   = wv >> 1;
  const int NT    = V_ >> 4;                 // 2000
  const int nt    = blockIdx.x * 4 + ntl;    // 500 blocks * 4 tiles
  v8f acc = {};
  for (int kt = 0; kt < 32; ++kt) {          // K = 1024
    v16bf a = load_frag(A + ((kt * 2 + mtile) << 9), lane);
    const unsigned short* bp = Bm + (((size_t)kt * NT + nt) << 9);
    __builtin_prefetch(bp + ((size_t)NT << 9), 0, 1);
    v16bf b = load_frag(bp, lane);
    acc = wmma_bf16(a, b, acc);
  }
  int n  = (nt << 4) + (lane & 15);
  int mb = ((lane >> 4) << 3) + (mtile << 4);
#pragma unroll
  for (int r = 0; r < 8; ++r)
    logits[(size_t)(mb + r) * V_ + n] = acc[r] + bias[n];
}

// ---------- vocab softmax + argmax (first-max tie-break, like jnp.argmax) ----------
__global__ __launch_bounds__(256) void probs_kernel(const float* __restrict__ logits,
                                                    float* __restrict__ probs_out,
                                                    int* __restrict__ argmax,
                                                    const int* __restrict__ flags, int t) {
  __shared__ float smax[256];
  __shared__ int   simax[256];
  __shared__ float ssum[256];
  const int b = blockIdx.x, tid = threadIdx.x;
  const float* row = logits + (size_t)b * V_;
  float mx = -3.0e38f; int mi = 0;
  for (int v = tid; v < V_; v += 256) {
    float x = row[v];
    if (x > mx) { mx = x; mi = v; }
  }
  smax[tid] = mx; simax[tid] = mi; __syncthreads();
  for (int off = 128; off; off >>= 1) {
    if (tid < off) {
      float o = smax[tid + off]; int oi = simax[tid + off];
      if (o > smax[tid] || (o == smax[tid] && oi < simax[tid])) { smax[tid] = o; simax[tid] = oi; }
    }
    __syncthreads();
  }
  float gmax = smax[0];
  float ls = 0.f;
  for (int v = tid; v < V_; v += 256) ls += __expf(row[v] - gmax);
  ssum[tid] = ls; __syncthreads();
  for (int off = 128; off; off >>= 1) { if (tid < off) ssum[tid] += ssum[tid + off]; __syncthreads(); }
  float scale = flags[0] ? 0.f : (1.f / ssum[0]);    // emit 0 after all sequences done
  float* out = probs_out + ((size_t)b * T_ + t) * V_;
  for (int v = tid; v < V_; v += 256) out[v] = __expf(row[v] - gmax) * scale;
  if (tid == 0) argmax[b] = simax[0];
}

// ---------- token/done state machine ----------
__global__ void finalize_kernel(float* __restrict__ tok_out, int* __restrict__ token,
                                int* __restrict__ done, int* __restrict__ flags,
                                const int* __restrict__ argmax, int t) {
  int b = threadIdx.x;                       // 32 threads, one wave32
  int a  = argmax[b];
  int dn = done[b] | (a == 2);               // END_TGT = 2
  int tk = dn ? 0 : a;                       // NULL_TGT = 0
  int emit = flags[0] ? 0 : tk;              // unwritten TensorArray -> 0
  tok_out[(size_t)b * T_ + t] = (float)emit;
  token[b] = tk;
  done[b]  = dn;
  unsigned long long bal = __ballot(dn != 0);
  if (b == 0 && (unsigned)(bal & 0xffffffffULL) == 0xffffffffu) flags[0] = 1;
}

// ---------- host ----------
extern "C" void kernel_launch(void* const* d_in, const int* in_sizes, int n_in,
                              void* d_out, int out_size, void* d_ws, size_t ws_size,
                              hipStream_t stream) {
  (void)in_sizes; (void)n_in; (void)out_size; (void)ws_size;
  const unsigned char* mask = (const unsigned char*)d_in[0];
  const float* enc = (const float*)d_in[1];
  const float* h_in[4] = {(const float*)d_in[2], (const float*)d_in[4],
                          (const float*)d_in[6], (const float*)d_in[8]};
  const float* c_in[4] = {(const float*)d_in[3], (const float*)d_in[5],
                          (const float*)d_in[7], (const float*)d_in[9]};
  const float* emb  = (const float*)d_in[11];
  const float* Wf[4] = {(const float*)d_in[12], (const float*)d_in[15],
                        (const float*)d_in[18], (const float*)d_in[21]};
  const float* Uf[4] = {(const float*)d_in[13], (const float*)d_in[16],
                        (const float*)d_in[19], (const float*)d_in[22]};
  const float* bv[4] = {(const float*)d_in[14], (const float*)d_in[17],
                        (const float*)d_in[20], (const float*)d_in[23]};
  const float* Wfc0 = (const float*)d_in[24];
  const float* bfc0 = (const float*)d_in[25];
  const float* Wfc1 = (const float*)d_in[26];
  const float* bfc1 = (const float*)d_in[27];

  float* tok_out  = (float*)d_out;
  float* prob_out = (float*)d_out + (size_t)B_ * T_;

  char* p = (char*)d_ws;
  auto alloc = [&](size_t bytes) -> char* {
    char* r = p; p += (bytes + 255) & ~(size_t)255; return r;
  };
  unsigned short *Wsw[4], *Usw[4];
  for (int i = 0; i < 4; ++i) Wsw[i] = (unsigned short*)alloc((size_t)H_ * 4 * H_ * 2);
  for (int i = 0; i < 4; ++i) Usw[i] = (unsigned short*)alloc((size_t)H_ * 4 * H_ * 2);
  unsigned short* fc0sw  = (unsigned short*)alloc((size_t)2 * H_ * H_ * 2);
  unsigned short* fc1sw  = (unsigned short*)alloc((size_t)H_ * V_ * 2);
  unsigned short* x_sw   = (unsigned short*)alloc((size_t)B_ * H_ * 2);
  unsigned short* cat_sw = (unsigned short*)alloc((size_t)B_ * 2 * H_ * 2);
  unsigned short* att_sw = (unsigned short*)alloc((size_t)B_ * H_ * 2);
  unsigned short* hsw[4][2];
  for (int i = 0; i < 4; ++i)
    for (int q = 0; q < 2; ++q) hsw[i][q] = (unsigned short*)alloc((size_t)B_ * H_ * 2);
  float *hws[4], *cws[4];
  for (int i = 0; i < 4; ++i) hws[i] = (float*)alloc((size_t)B_ * H_ * 4);
  for (int i = 0; i < 4; ++i) cws[i] = (float*)alloc((size_t)B_ * H_ * 4);
  float* scores = (float*)alloc((size_t)B_ * S_ * 4);
  float* watt   = (float*)alloc((size_t)B_ * S_ * 4);
  float* logits = (float*)alloc((size_t)B_ * V_ * 4);
  int* token = (int*)alloc(256);
  int* done  = (int*)alloc(256);
  int* amax  = (int*)alloc(256);
  int* flags = (int*)alloc(256);

  // one-time per call: bf16-convert + swizzle all weights (then 64 steps hit L2)
  for (int i = 0; i < 4; ++i) {
    swizzle_w_kernel<<<4096, 256, 0, stream>>>(Wf[i], Wsw[i], H_, 4 * H_);
    swizzle_w_kernel<<<4096, 256, 0, stream>>>(Uf[i], Usw[i], H_, 4 * H_);
  }
  swizzle_w_kernel<<<2048, 256, 0, stream>>>(Wfc0, fc0sw, 2 * H_, H_);
  swizzle_w_kernel<<<8192, 256, 0, stream>>>(Wfc1, fc1sw, H_, V_);
  for (int i = 0; i < 4; ++i)
    init_cell_kernel<<<128, 256, 0, stream>>>(h_in[i], c_in[i], hws[i], cws[i], hsw[i][0]);
  init_token_kernel<<<1, 64, 0, stream>>>(token, done, flags);

  for (int t = 0; t < T_; ++t) {
    const int rp = t & 1, wp = rp ^ 1;
    embed_kernel<<<128, 256, 0, stream>>>(emb, token, x_sw);
    lstm_cell_kernel<<<64, 256, 0, stream>>>(x_sw,       Wsw[0], hsw[0][rp], Usw[0], bv[0], cws[0], hws[0], hsw[0][wp]);
    lstm_cell_kernel<<<64, 256, 0, stream>>>(hsw[0][wp], Wsw[1], hsw[1][rp], Usw[1], bv[1], cws[1], hws[1], hsw[1][wp]);
    lstm_cell_kernel<<<64, 256, 0, stream>>>(hsw[1][wp], Wsw[2], hsw[2][rp], Usw[2], bv[2], cws[2], hws[2], hsw[2][wp]);
    lstm_cell_kernel<<<64, 256, 0, stream>>>(hsw[2][wp], Wsw[3], hsw[3][rp], Usw[3], bv[3], cws[3], hws[3], hsw[3][wp]);
    attn_scores_kernel<<<512, 256, 0, stream>>>(hws[3], enc, mask, scores);
    attn_softmax_kernel<<<32, 128, 0, stream>>>(scores, watt);
    context_cat_kernel<<<128, 256, 0, stream>>>(watt, enc, hws[3], cat_sw);
    fc0_kernel<<<16, 256, 0, stream>>>(cat_sw, fc0sw, bfc0, att_sw);
    fc1_kernel<<<500, 256, 0, stream>>>(att_sw, fc1sw, bfc1, logits);
    probs_kernel<<<32, 256, 0, stream>>>(logits, prob_out, amax, flags, t);
    finalize_kernel<<<1, 32, 0, stream>>>(tok_out, token, done, flags, amax, t);
  }
}
